// Attention_46969762349763
// MI455X (gfx1250) — matrix operand order repaired and checked by reading, not compiled, not run on hardware
//
#include <hip/hip_runtime.h>

// MI455X / gfx1250 attention forward, fused flash-style (P materialized since
// `attention` is a required output). HBM-bound on the 805MB attention write;
// all GEMM operands are L2-resident. Matmuls: v_wmma_f32_16x16x32_bf16.
// Attention stores use non-temporal hints (TH_NT) so the streaming 805MB
// write does not evict the L2-resident K/V working set.
//
// Pipeline:
//   cvt(x, qkv_w, proj_w -> bf16 in ws)          [NT reads, one-shot]
//   qkv:   -> Q(pre-scaled 1/8), K [B,H,N,64] bf16, V^T [B,H,64,N] bf16
//   fused: per (b,h,16-row block): S=Q@K^T, row softmax, P -> attention (NT fp32)
//          and P@V via LDS-staged bf16 fragments -> out_heads bf16
//   proj:  + bias -> out fp32

typedef __attribute__((ext_vector_type(16))) __bf16 v16bf;
typedef __attribute__((ext_vector_type(8)))  float  v8f;
typedef __attribute__((ext_vector_type(4)))  float  v4f;

union FragB16 {
    v16bf v;
    __bf16 h[16];
    uint4 q[2];
};

__device__ __forceinline__ unsigned short f2bfu(float f) {
    __bf16 t = (__bf16)f;                       // native RNE cvt
    return __builtin_bit_cast(unsigned short, t);
}

__device__ __forceinline__ v8f wmma_bf16(v16bf a, v16bf b, v8f c) {
    return __builtin_amdgcn_wmma_f32_16x16x32_bf16(
        false, a, false, b, (short)0, c, false, false);
}

// A (16x32 bf16 MxK): lane -> row m = l&15, half = l>>4;
// elem i -> K = k0 + 16*half + i  (2x 16B loads)
__device__ __forceinline__ v16bf load_a_bf16(const unsigned short* __restrict__ A,
                                             int lda, int m0, int k0, int lane) {
    int m = m0 + (lane & 15);
    int half = lane >> 4;
    const unsigned short* p = A + (size_t)m * lda + k0 + 16 * half;
    FragB16 fr;
    fr.q[0] = *(const uint4*)(p);
    fr.q[1] = *(const uint4*)(p + 8);
    return fr.v;
}

// B (32x16 bf16 KxN), memory Bm[n][k] (k contiguous per column n):
// lane -> col n = l&15, half = l>>4; elem i -> K = k0 + 16*half + i  (32 contiguous B)
__device__ __forceinline__ v16bf load_b_kc_bf16(const unsigned short* __restrict__ Bm,
                                                int ldb, int n0, int k0, int lane) {
    int n = n0 + (lane & 15);
    int half = lane >> 4;
    const unsigned short* p = Bm + (size_t)n * ldb + k0 + 16 * half;
    FragB16 fr;
    fr.q[0] = *(const uint4*)(p);
    fr.q[1] = *(const uint4*)(p + 8);
    return fr.v;
}

// ---- fp32 -> bf16 bulk convert (4 elems/thread, NT reads: source is one-shot)
__global__ __launch_bounds__(256) void cvt_bf16_kernel(const float* __restrict__ s,
                                                       unsigned short* __restrict__ d,
                                                       int n) {
    int i = (blockIdx.x * 256 + threadIdx.x) * 4;
    if (i + 3 < n) {
        v4f t = __builtin_nontemporal_load((const v4f*)(s + i));
        d[i + 0] = f2bfu(t.x); d[i + 1] = f2bfu(t.y);
        d[i + 2] = f2bfu(t.z); d[i + 3] = f2bfu(t.w);
    }
}

// ---- qkv GEMM: one 16x64 strip per wave (A reused across 4 WMMAs)
__global__ __launch_bounds__(256) void qkv_kernel(const unsigned short* __restrict__ xb,
                                                  const unsigned short* __restrict__ wb,
                                                  unsigned short* __restrict__ q,
                                                  unsigned short* __restrict__ k,
                                                  unsigned short* __restrict__ vt) {
    const int lane = threadIdx.x & 31;
    const int strip = blockIdx.x * 8 + (threadIdx.x >> 5);
    // strip = ((mat*12 + h) * 512 + m-tile): keeps (mat,h) derivable by shifts
    const int m0 = (strip & 511) * 16;
    const int mh = strip >> 9;                 // 0..35
    const int mat = mh / 12;
    const int hh  = mh % 12;
    const int n0 = mh * 64;
    v8f zero = {};
    v8f acc[4] = {zero, zero, zero, zero};
    for (int k0 = 0; k0 < 768; k0 += 32) {
        v16bf a = load_a_bf16(xb, 768, m0, k0, lane);
        #pragma unroll
        for (int j = 0; j < 4; ++j) {
            v16bf b = load_b_kc_bf16(wb, 768, n0 + j * 16, k0, lane);
            acc[j] = wmma_bf16(a, b, acc[j]);
        }
    }
    #pragma unroll
    for (int j = 0; j < 4; ++j) {
        const int d = j * 16 + (lane & 15);
        #pragma unroll
        for (int r = 0; r < 8; ++r) {
            const int m = m0 + r + 8 * (lane >> 4);
            const int bbn = m >> 11, nseq = m & 2047;
            const size_t bhh = (size_t)(bbn * 12 + hh);
            if (mat == 0)      q [(bhh * 2048 + nseq) * 64 + d] = f2bfu(acc[j][r] * 0.125f);
            else if (mat == 1) k [(bhh * 2048 + nseq) * 64 + d] = f2bfu(acc[j][r]);
            else               vt[(bhh * 64 + d) * 2048 + nseq] = f2bfu(acc[j][r]);
        }
    }
}

// ---- fused scores + softmax + attention store + P@V
// grid = B*H*(N/16) = 6144 blocks, 256 threads (8 waves).
// wave w owns score strip S[16 x 256] at cols [w*256, w*256+256).
__global__ __launch_bounds__(256) void attn_fused_kernel(const unsigned short* __restrict__ q,
                                                         const unsigned short* __restrict__ kmat,
                                                         const unsigned short* __restrict__ vt,
                                                         float* __restrict__ attn,
                                                         unsigned short* __restrict__ oh) {
    __shared__ float  red1[8][16];             // cross-wave row max
    __shared__ float  red2[8][16];             // cross-wave row sum
    __shared__ __bf16 pstage[8][16][32];       // per-wave P transpose stage (1KB/wave)
    __shared__ float  osum[16][64];            // P@V partial reduction (4KB)

    const int lane = threadIdx.x & 31;
    const int wave = threadIdx.x >> 5;
    const int half = lane >> 4;
    const int bh = blockIdx.x >> 7;
    const int m0 = (blockIdx.x & 127) * 16;
    const int bb = bh / 12, hh = bh % 12;
    const int nbase = wave * 256;

    const unsigned short* qb = q    + (size_t)bh * 2048 * 64;
    const unsigned short* kb = kmat + (size_t)bh * 2048 * 64;
    const unsigned short* vb = vt   + (size_t)bh * 64 * 2048;
    float* arow = attn + (size_t)bh * 2048 * 2048;

    for (int i = threadIdx.x; i < 16 * 64; i += 256) ((float*)osum)[i] = 0.f;
    __syncthreads();

    // ---- scores: 16 WMMA tiles per wave, Q fragment reused 16x (Q pre-scaled by 1/8)
    v8f zero = {};
    v8f s[16];
    #pragma unroll
    for (int j = 0; j < 16; ++j) s[j] = zero;
    #pragma unroll
    for (int k0 = 0; k0 < 64; k0 += 32) {
        v16bf a = load_a_bf16(qb, 64, m0, k0, lane);
        #pragma unroll
        for (int j = 0; j < 16; ++j) {
            v16bf b = load_b_kc_bf16(kb, 64, nbase + j * 16, k0, lane);
            s[j] = wmma_bf16(a, b, s[j]);
        }
    }

    // ---- row max: C-layout -> row (r + 8*half) lives in a 16-lane group at reg r
    float rmax[8], rsum[8];
    #pragma unroll
    for (int r = 0; r < 8; ++r) {
        float m = s[0][r];
        #pragma unroll
        for (int j = 1; j < 16; ++j) m = fmaxf(m, s[j][r]);
        #pragma unroll
        for (int off = 1; off < 16; off <<= 1) m = fmaxf(m, __shfl_xor(m, off, 32));
        rmax[r] = m;
    }
    if ((lane & 15) == 0) {
        #pragma unroll
        for (int r = 0; r < 8; ++r) red1[wave][r + 8 * half] = rmax[r];
    }
    __syncthreads();
    #pragma unroll
    for (int r = 0; r < 8; ++r) {
        float m = red1[0][r + 8 * half];
        #pragma unroll
        for (int w = 1; w < 8; ++w) m = fmaxf(m, red1[w][r + 8 * half]);
        rmax[r] = m;
    }

    // ---- exp + row sum
    #pragma unroll
    for (int r = 0; r < 8; ++r) {
        float acc = 0.f;
        #pragma unroll
        for (int j = 0; j < 16; ++j) {
            float e = __expf(s[j][r] - rmax[r]);
            s[j][r] = e;
            acc += e;
        }
        #pragma unroll
        for (int off = 1; off < 16; off <<= 1) acc += __shfl_xor(acc, off, 32);
        rsum[r] = acc;
    }
    if ((lane & 15) == 0) {
        #pragma unroll
        for (int r = 0; r < 8; ++r) red2[wave][r + 8 * half] = rsum[r];
    }
    __syncthreads();
    #pragma unroll
    for (int r = 0; r < 8; ++r) {
        float t = red2[0][r + 8 * half];
        #pragma unroll
        for (int w = 1; w < 8; ++w) t += red2[w][r + 8 * half];
        rsum[r] = 1.0f / t;
    }

    // ---- normalize; NT-store attention fp32 (write-once stream, keep L2 for K/V);
    //      transpose P via LDS; P@V WMMAs
    v8f oacc[4] = {zero, zero, zero, zero};
    for (int jp = 0; jp < 8; ++jp) {           // pairs of 16-col tiles -> K=32
        #pragma unroll
        for (int jj = 0; jj < 2; ++jj) {
            const int j = jp * 2 + jj;
            #pragma unroll
            for (int r = 0; r < 8; ++r) {
                const float p = s[j][r] * rsum[r];
                const int mrow = r + 8 * half;
                __builtin_nontemporal_store(
                    p, &arow[(size_t)(m0 + mrow) * 2048 + nbase + j * 16 + (lane & 15)]);
                pstage[wave][mrow][jj * 16 + (lane & 15)] = (__bf16)p;
            }
        }
        // wave-local LDS RAW: CDNA5 split-counter wait
        asm volatile("s_wait_dscnt 0" ::: "memory");
        v16bf a = load_a_bf16((const unsigned short*)&pstage[wave][0][0], 32, 0, 0, lane);
        #pragma unroll
        for (int jd = 0; jd < 4; ++jd) {       // all 64 output dims, B from L2 V^T
            v16bf b = load_b_kc_bf16(vb, 2048, jd * 16, nbase + jp * 32, lane);
            oacc[jd] = wmma_bf16(a, b, oacc[jd]);
        }
    }

    // ---- combine 8 wave partials, emit out_heads bf16 [B,N,C]
    #pragma unroll
    for (int jd = 0; jd < 4; ++jd) {
        #pragma unroll
        for (int r = 0; r < 8; ++r)
            atomicAdd(&osum[r + 8 * half][jd * 16 + (lane & 15)], oacc[jd][r]);
    }
    __syncthreads();
    const size_t obase = ((size_t)(bb * 2048 + m0)) * 768 + hh * 64;
    for (int i = threadIdx.x; i < 16 * 64; i += 256) {
        const int row = i >> 6, d = i & 63;
        oh[obase + (size_t)row * 768 + d] = f2bfu(osum[row][d]);
    }
}

// ---- projection GEMM + bias: one 16x64 strip per wave
__global__ __launch_bounds__(256) void proj_kernel(const unsigned short* __restrict__ oh,
                                                   const unsigned short* __restrict__ wb,
                                                   const float* __restrict__ bias,
                                                   float* __restrict__ out) {
    const int lane = threadIdx.x & 31;
    const int strip = blockIdx.x * 8 + (threadIdx.x >> 5);
    const int m0 = (strip / 12) * 16;          // 512 row tiles
    const int n0 = (strip % 12) * 64;          // 12 col strips
    v8f zero = {};
    v8f acc[4] = {zero, zero, zero, zero};
    for (int k0 = 0; k0 < 768; k0 += 32) {
        v16bf a = load_a_bf16(oh, 768, m0, k0, lane);
        #pragma unroll
        for (int j = 0; j < 4; ++j) {
            v16bf b = load_b_kc_bf16(wb, 768, n0 + j * 16, k0, lane);
            acc[j] = wmma_bf16(a, b, acc[j]);
        }
    }
    #pragma unroll
    for (int j = 0; j < 4; ++j) {
        const int col = n0 + j * 16 + (lane & 15);
        const float bv = bias[col];
        #pragma unroll
        for (int r = 0; r < 8; ++r) {
            const int m = m0 + r + 8 * (lane >> 4);
            out[(size_t)m * 768 + col] = acc[j][r] + bv;
        }
    }
}

extern "C" void kernel_launch(void* const* d_in, const int* in_sizes, int n_in,
                              void* d_out, int out_size, void* d_ws, size_t ws_size,
                              hipStream_t stream) {
    (void)in_sizes; (void)n_in; (void)out_size; (void)ws_size;
    const float* x      = (const float*)d_in[0];  // [4,2048,768]
    const float* qkv_w  = (const float*)d_in[1];  // [2304,768]
    const float* proj_w = (const float*)d_in[2];  // [768,768]
    const float* proj_b = (const float*)d_in[3];  // [768]

    float* out  = (float*)d_out;                  // [4,2048,768]
    float* attn = out + (size_t)4 * 2048 * 768;   // [4,12,2048,2048]

    // ws layout (ushort elements): q,k,vt,oh (6291456 each), xb, wqkv, wproj
    unsigned short* q   = (unsigned short*)d_ws;
    unsigned short* k   = q   + (size_t)6291456;
    unsigned short* vt  = k   + (size_t)6291456;
    unsigned short* oh  = vt  + (size_t)6291456;
    unsigned short* xb  = oh  + (size_t)6291456;
    unsigned short* wqk = xb  + (size_t)6291456;  // 2304*768
    unsigned short* wpr = wqk + (size_t)1769472;  // 768*768
    // total = 67,633,152 bytes

    cvt_bf16_kernel  <<<6144, 256, 0, stream>>>(x,      xb,  6291456);
    cvt_bf16_kernel  <<<1728, 256, 0, stream>>>(qkv_w,  wqk, 1769472);
    cvt_bf16_kernel  <<< 576, 256, 0, stream>>>(proj_w, wpr, 589824);

    qkv_kernel       <<<2304, 256, 0, stream>>>(xb, wqk, q, k, vt);   // 36*512 strips /8
    attn_fused_kernel<<<6144, 256, 0, stream>>>(q, k, vt, attn, oh);  // 48*128 row blocks
    proj_kernel      <<< 768, 256, 0, stream>>>(oh, wpr, proj_b, out);// 512*12 strips /8
}
